// UniversalDecoderLayer_62345745269551
// MI455X (gfx1250) — compile-verified, matrix operand order checked by reading, not run on hardware
//
#include <hip/hip_runtime.h>
#include <hip/hip_bf16.h>

// ---- problem constants ----
#define BB   2
#define SS   512
#define DDIM 1024
#define HH   16
#define DHH  64
#define FF   4096
#define BSTOK (BB*SS)                       // 1024 tokens
#define BSD   ((long)BB*SS*DDIM)            // 1,048,576 elems
#define NHEADS (BB*HH)                      // 32
#define THRESH 0.8f
#define NEGBIG 1e10f

typedef __bf16 bf16;
typedef __bf16 bf16x16 __attribute__((ext_vector_type(16)));
typedef float  floatx8 __attribute__((ext_vector_type(8)));

// ---- CDNA5 async global->LDS helpers (ASYNCcnt path, ISA ch.10 async ops) ----
__device__ __forceinline__ void async_ld_b128(void* lds_dst, const void* gsrc) {
    unsigned l = (unsigned)(size_t)lds_dst;           // low 32 bits of generic ptr == LDS offset
    asm volatile("global_load_async_to_lds_b128 %0, %1, off"
                 :: "v"(l), "v"(gsrc) : "memory");
}
__device__ __forceinline__ void wait_async0() {
    asm volatile("s_wait_asynccnt 0x0" ::: "memory");
}

// =====================================================================
// Generic batched WMMA GEMM:  C[z] = A[z](MxK,bf16) * B[z](bf16) (+bias +resid)
// transB==0: B is KxN row-major; transB==1: B is NxK row-major (C = A*B^T).
// Block tile 128x128, BK=32, 256 threads = 8 waves (2x4 wave grid),
// each wave computes a 64x32 patch = 4x2 WMMA tiles of 16x16.
// LDS tiles are double-buffered; A and B^T tiles stream in via
// global_load_async_to_lds_b128 overlapped with the WMMA pipe.
// =====================================================================
__global__ __launch_bounds__(256) void gemm_wmma_kernel(
    const bf16* __restrict__ A, const bf16* __restrict__ Bm,
    const float* __restrict__ bias, const float* __restrict__ resid,
    float* __restrict__ C, int M, int N, int K, int transB,
    long sA, long sB, long sC)
{
    __shared__ bf16 As[2][128][32];
    __shared__ bf16 Bs[2][128][32];   // stored [n][k]  (i.e. B^T tile)

    const int z = blockIdx.z;
    A  += (long)z * sA;
    Bm += (long)z * sB;
    C  += (long)z * sC;

    const int bm   = blockIdx.y * 128;
    const int bn   = blockIdx.x * 128;
    const int tid  = threadIdx.x;
    const int wave = tid >> 5;
    const int lane = tid & 31;
    const int wm   = (wave & 1) * 64;       // 2 waves along M
    const int wn   = (wave >> 1) * 32;      // 4 waves along N
    const int lmod = lane & 15;
    const int lhi  = (lane >> 4) & 1;       // which half-wave

    floatx8 acc[4][2];
#pragma unroll
    for (int i = 0; i < 4; ++i)
#pragma unroll
        for (int j = 0; j < 2; ++j) acc[i][j] = (floatx8)(0.0f);

    // stage K-tile k0 into LDS buffer `buf` (async for contiguous paths)
    auto stage = [&](int buf, int k0) {
        {   // A tile: 128 x 32, each thread DMAs 2x16B directly into LDS
            int r  = tid >> 1;
            int cp = (tid & 1) * 16;
            const bf16* src = A + (long)(bm + r) * K + k0 + cp;
            async_ld_b128(&As[buf][r][cp],     src);
            async_ld_b128(&As[buf][r][cp + 8], src + 8);
        }
        if (transB) {
            int r  = tid >> 1;
            int cp = (tid & 1) * 16;
            if (bn + r < N) {
                const bf16* src = Bm + (long)(bn + r) * K + k0 + cp;
                async_ld_b128(&Bs[buf][r][cp],     src);
                async_ld_b128(&Bs[buf][r][cp + 8], src + 8);
            } else {
                *(uint4*)&Bs[buf][r][cp]     = make_uint4(0u,0u,0u,0u);
                *(uint4*)&Bs[buf][r][cp + 8] = make_uint4(0u,0u,0u,0u);
            }
        } else {
            // transpose-scatter: must go through VGPRs (tracked DS stores)
            int k  = tid >> 3;          // 0..31
            int n0 = (tid & 7) * 16;    // 0..112
            bf16 tmp[16];
            if (bn + n0 < N) {
                const bf16* src = Bm + (long)(k0 + k) * N + bn + n0;
                *(uint4*)&tmp[0] = *(const uint4*)src;
                *(uint4*)&tmp[8] = *(const uint4*)(src + 8);
            } else {
#pragma unroll
                for (int j = 0; j < 16; ++j) tmp[j] = (bf16)0.0f;
            }
#pragma unroll
            for (int j = 0; j < 16; ++j) Bs[buf][n0 + j][k] = tmp[j];
        }
    };

    const int nk = K >> 5;
    stage(0, 0);

    for (int it = 0; it < nk; ++it) {
        const int cur = it & 1;
        wait_async0();          // own wave's async DMAs into buf[cur] done
        __syncthreads();        // all waves' writes to buf[cur] visible

        if (it + 1 < nk) stage(cur ^ 1, (it + 1) << 5);   // prefetch next tile

        // ---- fragments + WMMA on buf[cur] ----
        bf16x16 bfrag[2];
#pragma unroll
        for (int j = 0; j < 2; ++j) {
            const bf16* bp = &Bs[cur][wn + j * 16 + lmod][lhi ? 8 : 0];
            uint4* d = (uint4*)&bfrag[j];
            d[0] = *(const uint4*)bp;
            d[1] = *(const uint4*)(bp + 16);
        }
#pragma unroll
        for (int i = 0; i < 4; ++i) {
            bf16x16 afrag;
            const bf16* ap = &As[cur][wm + i * 16 + lmod][lhi ? 8 : 0];
            uint4* d = (uint4*)&afrag;
            d[0] = *(const uint4*)ap;
            d[1] = *(const uint4*)(ap + 16);
#pragma unroll
            for (int j = 0; j < 2; ++j)
                acc[i][j] = __builtin_amdgcn_wmma_f32_16x16x32_bf16(
                    false, afrag, false, bfrag[j], (short)0, acc[i][j], false, false);
        }
        __syncthreads();        // all waves done reading buf[cur] before it is overwritten
    }

    // ---- epilogue: bias + residual, fp32 store ----
#pragma unroll
    for (int i = 0; i < 4; ++i) {
#pragma unroll
        for (int j = 0; j < 2; ++j) {
            int cm = bm + wm + i * 16 + lhi * 8;
            int cn = bn + wn + j * 16 + lmod;
            if (cn < N) {
                float bv = bias ? bias[cn] : 0.0f;
#pragma unroll
                for (int r = 0; r < 8; ++r) {
                    float v = acc[i][j][r] + bv;
                    long idx = (long)(cm + r) * N + cn;
                    if (resid) v += resid[idx];
                    C[idx] = v;
                }
            }
        }
    }
}

// ==================== LayerNorm (one 256-thread block / row, D=1024) ====================
__global__ __launch_bounds__(256) void layernorm_kernel(
    const float* __restrict__ x, const float* __restrict__ w,
    const float* __restrict__ b, bf16* __restrict__ out)
{
    __shared__ float red[256];
    const int row = blockIdx.x;
    const float* xr = x + (long)row * DDIM;
    float s = 0.f;
    for (int i = threadIdx.x; i < DDIM; i += 256) s += xr[i];
    red[threadIdx.x] = s; __syncthreads();
    for (int off = 128; off > 0; off >>= 1) {
        if (threadIdx.x < off) red[threadIdx.x] += red[threadIdx.x + off];
        __syncthreads();
    }
    float mean = red[0] * (1.0f / DDIM); __syncthreads();
    float v = 0.f;
    for (int i = threadIdx.x; i < DDIM; i += 256) { float d = xr[i] - mean; v += d * d; }
    red[threadIdx.x] = v; __syncthreads();
    for (int off = 128; off > 0; off >>= 1) {
        if (threadIdx.x < off) red[threadIdx.x] += red[threadIdx.x + off];
        __syncthreads();
    }
    float inv = rsqrtf(red[0] * (1.0f / DDIM) + 1e-12f);
    bf16* o = out + (long)row * DDIM;
    for (int i = threadIdx.x; i < DDIM; i += 256)
        o[i] = (bf16)((xr[i] - mean) * inv * w[i] + b[i]);
}

// ==================== row softmax over S=512, one wave / row ====================
__global__ __launch_bounds__(256) void softmax_kernel(
    const float* __restrict__ scores, bf16* __restrict__ probs, int causal)
{
    int gid  = blockIdx.x * 8 + (threadIdx.x >> 5);   // row id in [0, NHEADS*S)
    int lane = threadIdx.x & 31;
    int q = gid & (SS - 1);
    const float* row = scores + (long)gid * SS;
    float vals[16];
    float mx = -3.4e38f;
#pragma unroll
    for (int i = 0; i < 16; ++i) {
        int c = lane + i * 32;
        float v = row[c];
        if (causal && c > q) v = -NEGBIG;
        vals[i] = v;
        mx = fmaxf(mx, v);
    }
    for (int off = 16; off > 0; off >>= 1) mx = fmaxf(mx, __shfl_xor(mx, off));
    float sum = 0.f;
#pragma unroll
    for (int i = 0; i < 16; ++i) { vals[i] = __expf(vals[i] - mx); sum += vals[i]; }
    for (int off = 16; off > 0; off >>= 1) sum += __shfl_xor(sum, off);
    float inv = 1.0f / sum;
    bf16* o = probs + (long)gid * SS;
#pragma unroll
    for (int i = 0; i < 16; ++i) o[lane + i * 32] = (bf16)(vals[i] * inv);
}

// ==================== layout / elementwise kernels ====================
__global__ void f2bf_kernel(const float* __restrict__ in, bf16* __restrict__ out, long n) {
    long i = (long)blockIdx.x * 256 + threadIdx.x;
    if (i < n) out[i] = (bf16)in[i];
}

__global__ void split_qkv_kernel(const float* __restrict__ qkv,
                                 bf16* __restrict__ q, bf16* __restrict__ k, bf16* __restrict__ v) {
    long i = (long)blockIdx.x * 256 + threadIdx.x;
    if (i >= BSD) return;
    int d = (int)(i % DDIM);
    long bs = i / DDIM;
    int s = (int)(bs % SS), b = (int)(bs / SS);
    long o = (((long)b * HH + d / DHH) * SS + s) * DHH + (d % DHH);
    const float* base = qkv + bs * 3 * DDIM;
    q[o] = (bf16)base[d];
    k[o] = (bf16)base[DDIM + d];
    v[o] = (bf16)base[2 * DDIM + d];
}

__global__ void split_heads_kernel(const float* __restrict__ in, bf16* __restrict__ out) {
    long i = (long)blockIdx.x * 256 + threadIdx.x;
    if (i >= BSD) return;
    int d = (int)(i % DDIM);
    long bs = i / DDIM;
    int s = (int)(bs % SS), b = (int)(bs / SS);
    out[(((long)b * HH + d / DHH) * SS + s) * DHH + (d % DHH)] = (bf16)in[i];
}

__global__ void merge_heads_kernel(const float* __restrict__ in, bf16* __restrict__ out) {
    long i = (long)blockIdx.x * 256 + threadIdx.x;
    if (i >= BSD) return;
    int d = (int)(i % DDIM);
    long bs = i / DDIM;
    int s = (int)(bs % SS), b = (int)(bs / SS);
    out[i] = (bf16)in[(((long)b * HH + d / DHH) * SS + s) * DHH + (d % DHH)];
}

__global__ void gelu_kernel(const float* __restrict__ in, bf16* __restrict__ out, long n) {
    long i = (long)blockIdx.x * 256 + threadIdx.x;
    if (i >= n) return;
    float u = in[i];
    float t = tanhf(0.7978845608028654f * (u + 0.044715f * u * u * u));
    out[i] = (bf16)(0.5f * u * (1.0f + t));
}

// ==================== ACT machinery ====================
__global__ void init_state_kernel(const float* __restrict__ x, float* __restrict__ xs,
                                  float* __restrict__ hp, float* __restrict__ rem,
                                  float* __restrict__ nup) {
    long i = (long)blockIdx.x * 256 + threadIdx.x;
    if (i < BSD) xs[i] = x[i];
    if (i < BSTOK) { hp[i] = 0.f; rem[i] = 0.f; nup[i] = 0.f; }
}

__global__ void cond_kernel(const float* __restrict__ hp, const float* __restrict__ nup,
                            int* __restrict__ flag) {
    __shared__ int anyv;
    if (threadIdx.x == 0) anyv = 0;
    __syncthreads();
    int i = threadIdx.x;                // 1024 threads == BSTOK
    if ((hp[i] < THRESH) && (nup[i] < 8.0f)) atomicOr(&anyv, 1);
    __syncthreads();
    if (threadIdx.x == 0) *flag = anyv;
}

// p = sigmoid(xs @ prob_w + prob_b), one wave / token
__global__ __launch_bounds__(256) void prob_kernel(const float* __restrict__ xs,
                                                   const float* __restrict__ pw,
                                                   const float* __restrict__ pb,
                                                   float* __restrict__ p) {
    int tok  = blockIdx.x * 8 + (threadIdx.x >> 5);
    int lane = threadIdx.x & 31;
    const float* xr = xs + (long)tok * DDIM;
    float s = 0.f;
    for (int i = lane; i < DDIM; i += 32) s += xr[i] * pw[i];
    for (int off = 16; off > 0; off >>= 1) s += __shfl_xor(s, off);
    if (lane == 0) p[tok] = 1.0f / (1.0f + __expf(-(s + pb[0])));
}

__global__ void act_update_kernel(const float* __restrict__ p, float* __restrict__ hp,
                                  float* __restrict__ rem, float* __restrict__ nup,
                                  const int* __restrict__ flag) {
    int i = blockIdx.x * 256 + threadIdx.x;
    if (i >= BSTOK) return;
    float hpv = hp[i], remv = rem[i], pv = p[i];
    float still      = (hpv < 1.0f) ? 1.f : 0.f;
    float hp_try     = hpv + pv * still;
    float new_halted = ((hp_try > THRESH) ? 1.f : 0.f) * still;
    float still2     = ((hp_try <= THRESH) ? 1.f : 0.f) * still;
    float hp1        = hpv + pv * still2;
    float rem1       = remv + new_halted * (1.0f - hp1);
    float hp2        = hp1 + new_halted * rem1;
    float nup1       = still2 + new_halted;
    if (*flag) { hp[i] = hp2; rem[i] = rem1; nup[i] = nup1; }
}

__global__ void select_x_kernel(const float* __restrict__ x3, float* __restrict__ xs,
                                const int* __restrict__ flag) {
    if (!*flag) return;
    long i = (long)blockIdx.x * 256 + threadIdx.x;
    if (i < BSD) xs[i] = x3[i];
}

__global__ void output_kernel(const float* __restrict__ xs, const float* __restrict__ nup,
                              const float* __restrict__ rem, float* __restrict__ out) {
    long i = (long)blockIdx.x * 256 + threadIdx.x;
    if (i < BSD) out[i] = xs[i];
    else if (i < BSD + BSTOK) out[i] = nup[i - BSD];
    else if (i < BSD + 2 * BSTOK) out[i] = rem[i - BSD - BSTOK];
}

// ==================== host orchestration ====================
static void launch_gemm(const bf16* A, const bf16* Bm, const float* bias, const float* resid,
                        float* C, int M, int N, int K, int transB, int batch,
                        long sA, long sB, long sC, hipStream_t stream) {
    dim3 grid((N + 127) / 128, M / 128, batch);
    gemm_wmma_kernel<<<grid, 256, 0, stream>>>(A, Bm, bias, resid, C, M, N, K, transB, sA, sB, sC);
}

static inline dim3 elgrid(long n) { return dim3((unsigned)((n + 255) / 256)); }

extern "C" void kernel_launch(void* const* d_in, const int* in_sizes, int n_in,
                              void* d_out, int out_size, void* d_ws, size_t ws_size,
                              hipStream_t stream) {
    const float* x      = (const float*)d_in[0];
    const float* enc    = (const float*)d_in[1];
    const float* ln1_w  = (const float*)d_in[2];
    const float* ln1_b  = (const float*)d_in[3];
    const float* ln2_w  = (const float*)d_in[4];
    const float* ln2_b  = (const float*)d_in[5];
    const float* ln3_w  = (const float*)d_in[6];
    const float* ln3_b  = (const float*)d_in[7];
    const float* qkv_w  = (const float*)d_in[8];
    const float* qkv_b  = (const float*)d_in[9];
    const float* aproj_w= (const float*)d_in[10];
    const float* aproj_b= (const float*)d_in[11];
    const float* wq_w   = (const float*)d_in[12];
    const float* wq_b   = (const float*)d_in[13];
    const float* wk_w   = (const float*)d_in[14];
    const float* wk_b   = (const float*)d_in[15];
    const float* wv_w   = (const float*)d_in[16];
    const float* wv_b   = (const float*)d_in[17];
    const float* cproj_w= (const float*)d_in[18];
    const float* cproj_b= (const float*)d_in[19];
    const float* fc_w   = (const float*)d_in[20];
    const float* fc_b   = (const float*)d_in[21];
    const float* mproj_w= (const float*)d_in[22];
    const float* mproj_b= (const float*)d_in[23];
    const float* prob_w = (const float*)d_in[24];
    const float* prob_b = (const float*)d_in[25];

    char* ws = (char*)d_ws;
    size_t off = 0;
    auto alloc = [&](size_t bytes) -> void* {
        void* p = ws + off;
        off += (bytes + 255) & ~(size_t)255;
        return p;
    };

    float* xs    = (float*)alloc(BSD * 4);
    float* x1    = (float*)alloc(BSD * 4);
    float* x2    = (float*)alloc(BSD * 4);
    float* x3    = (float*)alloc(BSD * 4);
    bf16*  lnb   = (bf16*) alloc(BSD * 2);
    float* qkvb  = (float*)alloc((size_t)BSTOK * 3 * DDIM * 4);      // also qc fp32 temp
    bf16*  qh    = (bf16*) alloc(BSD * 2);
    bf16*  kh    = (bf16*) alloc(BSD * 2);
    bf16*  vh    = (bf16*) alloc(BSD * 2);
    bf16*  kench = (bf16*) alloc(BSD * 2);
    bf16*  vench = (bf16*) alloc(BSD * 2);
    float* scores= (float*)alloc((size_t)NHEADS * SS * SS * 4);      // also fc fp32 (16MB of 32MB)
    bf16*  probs = (bf16*) alloc((size_t)NHEADS * SS * SS * 2);      // also gelu bf16 (8MB of 16MB)
    float* attnb = (float*)alloc((size_t)NHEADS * SS * DHH * 4);
    bf16*  merged= (bf16*) alloc(BSD * 2);
    bf16*  w_qkv = (bf16*) alloc((size_t)DDIM * 3 * DDIM * 2);
    bf16*  w_ap  = (bf16*) alloc((size_t)DDIM * DDIM * 2);
    bf16*  w_q   = (bf16*) alloc((size_t)DDIM * DDIM * 2);
    bf16*  w_k   = (bf16*) alloc((size_t)DDIM * DDIM * 2);
    bf16*  w_v   = (bf16*) alloc((size_t)DDIM * DDIM * 2);
    bf16*  w_cp  = (bf16*) alloc((size_t)DDIM * DDIM * 2);
    bf16*  w_fc  = (bf16*) alloc((size_t)DDIM * FF * 2);
    bf16*  w_mp  = (bf16*) alloc((size_t)FF * DDIM * 2);
    float* pbuf  = (float*)alloc(BSTOK * 4);
    float* hp    = (float*)alloc(BSTOK * 4);
    float* rem   = (float*)alloc(BSTOK * 4);
    float* nup   = (float*)alloc(BSTOK * 4);
    int*   flag  = (int*)  alloc(256);

    // ---- convert weights fp32 -> bf16 ----
    f2bf_kernel<<<elgrid((long)DDIM*3*DDIM), 256, 0, stream>>>(qkv_w, w_qkv, (long)DDIM*3*DDIM);
    f2bf_kernel<<<elgrid((long)DDIM*DDIM),   256, 0, stream>>>(aproj_w, w_ap, (long)DDIM*DDIM);
    f2bf_kernel<<<elgrid((long)DDIM*DDIM),   256, 0, stream>>>(wq_w, w_q, (long)DDIM*DDIM);
    f2bf_kernel<<<elgrid((long)DDIM*DDIM),   256, 0, stream>>>(wk_w, w_k, (long)DDIM*DDIM);
    f2bf_kernel<<<elgrid((long)DDIM*DDIM),   256, 0, stream>>>(wv_w, w_v, (long)DDIM*DDIM);
    f2bf_kernel<<<elgrid((long)DDIM*DDIM),   256, 0, stream>>>(cproj_w, w_cp, (long)DDIM*DDIM);
    f2bf_kernel<<<elgrid((long)DDIM*FF),     256, 0, stream>>>(fc_w, w_fc, (long)DDIM*FF);
    f2bf_kernel<<<elgrid((long)FF*DDIM),     256, 0, stream>>>(mproj_w, w_mp, (long)FF*DDIM);

    // ---- precompute cross-attn K/V from ln2(enc) ----
    layernorm_kernel<<<BSTOK, 256, 0, stream>>>(enc, ln2_w, ln2_b, lnb);
    launch_gemm(lnb, w_k, wk_b, nullptr, qkvb, BSTOK, DDIM, DDIM, 0, 1, 0, 0, 0, stream);
    split_heads_kernel<<<elgrid(BSD), 256, 0, stream>>>(qkvb, kench);
    launch_gemm(lnb, w_v, wv_b, nullptr, qkvb, BSTOK, DDIM, DDIM, 0, 1, 0, 0, 0, stream);
    split_heads_kernel<<<elgrid(BSD), 256, 0, stream>>>(qkvb, vench);

    // ---- init ACT state ----
    init_state_kernel<<<elgrid(BSD), 256, 0, stream>>>(x, xs, hp, rem, nup);

    const long sQK = (long)SS * DHH, sSC = (long)SS * SS;

    for (int step = 0; step < 8; ++step) {
        // halting bookkeeping (cond from PRE-update hp/nup)
        cond_kernel<<<1, 1024, 0, stream>>>(hp, nup, flag);
        prob_kernel<<<BSTOK / 8, 256, 0, stream>>>(xs, prob_w, prob_b, pbuf);
        act_update_kernel<<<(BSTOK + 255) / 256, 256, 0, stream>>>(pbuf, hp, rem, nup, flag);

        // ---- self attention ----
        layernorm_kernel<<<BSTOK, 256, 0, stream>>>(xs, ln1_w, ln1_b, lnb);
        launch_gemm(lnb, w_qkv, qkv_b, nullptr, qkvb, BSTOK, 3 * DDIM, DDIM, 0, 1, 0, 0, 0, stream);
        split_qkv_kernel<<<elgrid(BSD), 256, 0, stream>>>(qkvb, qh, kh, vh);
        launch_gemm(qh, kh, nullptr, nullptr, scores, SS, SS, DHH, 1, NHEADS, sQK, sQK, sSC, stream);
        softmax_kernel<<<(NHEADS * SS) / 8, 256, 0, stream>>>(scores, probs, 1);
        launch_gemm(probs, vh, nullptr, nullptr, attnb, SS, DHH, SS, 0, NHEADS, sSC, sQK, sQK, stream);
        merge_heads_kernel<<<elgrid(BSD), 256, 0, stream>>>(attnb, merged);
        launch_gemm(merged, w_ap, aproj_b, xs, x1, BSTOK, DDIM, DDIM, 0, 1, 0, 0, 0, stream);

        // ---- cross attention ----
        layernorm_kernel<<<BSTOK, 256, 0, stream>>>(x1, ln2_w, ln2_b, lnb);
        launch_gemm(lnb, w_q, wq_b, nullptr, qkvb, BSTOK, DDIM, DDIM, 0, 1, 0, 0, 0, stream);
        split_heads_kernel<<<elgrid(BSD), 256, 0, stream>>>(qkvb, qh);
        launch_gemm(qh, kench, nullptr, nullptr, scores, SS, SS, DHH, 1, NHEADS, sQK, sQK, sSC, stream);
        softmax_kernel<<<(NHEADS * SS) / 8, 256, 0, stream>>>(scores, probs, 0);
        launch_gemm(probs, vench, nullptr, nullptr, attnb, SS, DHH, SS, 0, NHEADS, sSC, sQK, sQK, stream);
        merge_heads_kernel<<<elgrid(BSD), 256, 0, stream>>>(attnb, merged);
        launch_gemm(merged, w_cp, cproj_b, x1, x2, BSTOK, DDIM, DDIM, 0, 1, 0, 0, 0, stream);

        // ---- MLP ----
        layernorm_kernel<<<BSTOK, 256, 0, stream>>>(x2, ln3_w, ln3_b, lnb);
        launch_gemm(lnb, w_fc, fc_b, nullptr, scores /*fc fp32*/, BSTOK, FF, DDIM, 0, 1, 0, 0, 0, stream);
        gelu_kernel<<<elgrid((long)BSTOK * FF), 256, 0, stream>>>(scores, (bf16*)probs, (long)BSTOK * FF);
        launch_gemm((bf16*)probs, w_mp, mproj_b, x2, x3, BSTOK, DDIM, FF, 0, 1, 0, 0, 0, stream);

        // ---- conditional carry update ----
        select_x_kernel<<<elgrid(BSD), 256, 0, stream>>>(x3, xs, flag);
    }

    output_kernel<<<elgrid(BSD + 2 * BSTOK), 256, 0, stream>>>(xs, nup, rem, (float*)d_out);
}
